// RzLinear_18013092839692
// MI455X (gfx1250) — matrix operand order repaired and checked by reading, not compiled, not run on hardware
//
#include <hip/hip_runtime.h>

typedef __attribute__((ext_vector_type(16))) __bf16          v16bf;
typedef __attribute__((ext_vector_type(8)))  unsigned short  v8us;
typedef __attribute__((ext_vector_type(8)))  float           v8f;

// Problem constants (mirror the reference)
constexpr int       M_TOK   = 512;
constexpr int       KDIM    = 8192;
constexpr int       NDIM    = 8192;
constexpr int       HASHSZ  = 1 << 22;
constexpr long long PRIME   = 2038074743LL;
constexpr int       BK      = 32;     // K step == hash tile K
constexpr int       TILE_M  = 128;
constexpr int       TILE_N  = 128;
constexpr int       LSTR    = 40;     // LDS row stride (bf16 elems): 32 data + 8 pad, keeps 16B alignment

__device__ __forceinline__ unsigned short bf16_rn(float f) {
    unsigned int u = __builtin_bit_cast(unsigned int, f);
    u += 0x7FFFu + ((u >> 16) & 1u);            // round-to-nearest-even
    return (unsigned short)(u >> 16);
}
__device__ __forceinline__ float bf16_f32(unsigned short h) {
    unsigned int u = ((unsigned int)h) << 16;
    return __builtin_bit_cast(float, u);
}
__device__ __forceinline__ void split_store(float f, unsigned short* ph, unsigned short* pl) {
    unsigned short hb = bf16_rn(f);
    *ph = hb;
    *pl = bf16_rn(f - bf16_f32(hb));
}

union FragU {
    v16bf v;
    v8us  h[2];
};

__global__ __launch_bounds__(256) void rz_linear_wmma_bf16x3(
    const float* __restrict__ x,
    const float* __restrict__ hw,
    const float* __restrict__ bias,
    const long long* __restrict__ rnd,
    float* __restrict__ out)
{
    __shared__ unsigned short sAhi[TILE_M * LSTR];   // x tile, hi bf16, [m][k]
    __shared__ unsigned short sAlo[TILE_M * LSTR];   // x tile, lo bf16
    __shared__ unsigned short sBhi[TILE_N * LSTR];   // W tile transposed, hi bf16, [n][k]
    __shared__ unsigned short sBlo[TILE_N * LSTR];   // W tile transposed, lo bf16

    const int t    = threadIdx.x;
    const int lane = t & 31;
    const int wv   = t >> 5;                  // wave id 0..7
    const int n0   = blockIdx.x * TILE_N;
    const int m0   = blockIdx.y * TILE_M;

    const long long R1 = rnd[1];
    const long long R2 = rnd[2];
    const long long R3 = rnd[3];

    // Wave sub-tile placement: 2 waves over M (64 rows each), 4 over N (32 cols each)
    const int wm = (wv & 1) * 64;
    const int wn = (wv >> 1) * 32;
    const int mr = lane & 15;                 // row/col within 16-wide fragment
    const int ks = lane >> 4;                 // K-half selector
    const int ak = ks * 8;                    // A frag K base (elems 0..7 -> k0..k0+7, 8..15 -> k0+16..)
    const int bk = ks * 8;                    // B frag K base (elems 0..7 -> k0..k0+7, 8..15 -> k0+16..)

    // A staging geometry: 8 float4 per 32-wide row, 32 rows per pass
    const int kq = (t & 7) * 4;
    const int mb = t >> 3;

    // B staging geometry: 64 threads per 32x32 hashed block, wave-uniform block id
    const int       bblk = t >> 6;            // 0..3
    const int       l64  = t & 63;
    const long long ntg  = (long long)((n0 >> 5) + bblk);

    const int NKB = KDIM / BK;

    // ---------- software-pipeline registers: global loads one K-block ahead ----------
    float4 rA[4];
    float4 rB[4];

    auto load_global = [&](int kb) {
        const float* xsrc = x + (long long)(m0 + mb) * KDIM + kb * BK + kq;
#pragma unroll
        for (int p = 0; p < 4; ++p)
            rA[p] = *(const float4*)(xsrc + (long long)p * 32 * KDIM);

        long long hs    = (((long long)kb) * R3 + ntg * R2 + R1) % PRIME;
        int       start = (int)(hs % (long long)(HASHSZ - BK * 32 + 1));
#pragma unroll
        for (int p = 0; p < 4; ++p)
            rB[p] = *(const float4*)(hw + start + l64 * 4 + p * 256);
    };

    v8f acc[4][2];
#pragma unroll
    for (int mi = 0; mi < 4; ++mi)
#pragma unroll
        for (int ni = 0; ni < 2; ++ni)
            acc[mi][ni] = (v8f)0.0f;

    load_global(0);   // prologue

    for (int kb = 0; kb < NKB; ++kb) {
        __syncthreads();   // previous compute phase done reading LDS

        // ---------------- convert + store staged registers into LDS ----------------
#pragma unroll
        for (int p = 0; p < 4; ++p) {                 // A: [m][k], packed ushort4 stores
            const int m = mb + p * 32;
            float4 v = rA[p];
            ushort4 h4, l4;
            h4.x = bf16_rn(v.x); l4.x = bf16_rn(v.x - bf16_f32(h4.x));
            h4.y = bf16_rn(v.y); l4.y = bf16_rn(v.y - bf16_f32(h4.y));
            h4.z = bf16_rn(v.z); l4.z = bf16_rn(v.z - bf16_f32(h4.z));
            h4.w = bf16_rn(v.w); l4.w = bf16_rn(v.w - bf16_f32(h4.w));
            *(ushort4*)&sAhi[m * LSTR + kq] = h4;
            *(ushort4*)&sAlo[m * LSTR + kq] = l4;
        }
#pragma unroll
        for (int p = 0; p < 4; ++p) {                 // B: transpose into [n][k]
            const int i = l64 * 4 + p * 256;          // flat idx in 32x32 block (k*32+n)
            const int k = i >> 5;
            const int n = i & 31;
            const int col = bblk * 32 + n;
            float4 g = rB[p];
            split_store(g.x, &sBhi[(col + 0) * LSTR + k], &sBlo[(col + 0) * LSTR + k]);
            split_store(g.y, &sBhi[(col + 1) * LSTR + k], &sBlo[(col + 1) * LSTR + k]);
            split_store(g.z, &sBhi[(col + 2) * LSTR + k], &sBlo[(col + 2) * LSTR + k]);
            split_store(g.w, &sBhi[(col + 3) * LSTR + k], &sBlo[(col + 3) * LSTR + k]);
        }

        __syncthreads();

        // Issue next K-block's global loads now; latency hidden under the WMMAs below.
        if (kb + 1 < NKB)
            load_global(kb + 1);
        if (kb + 2 < NKB) {                           // distance-2 prefetch of x rows
            const float* xn = x + (long long)(m0 + mb) * KDIM + (kb + 2) * BK + kq;
#pragma unroll
            for (int p = 0; p < 4; ++p)
                __builtin_prefetch((const void*)(xn + (long long)p * 32 * KDIM), 0, 0);
        }

        // ---------------- compute: 8 tiles x 3 bf16 WMMAs ----------------
        FragU ah[4], al[4], bh[2], bl[2];
#pragma unroll
        for (int mi = 0; mi < 4; ++mi) {
            const unsigned short* rh = &sAhi[(wm + mi * 16 + mr) * LSTR + ak];
            const unsigned short* rl = &sAlo[(wm + mi * 16 + mr) * LSTR + ak];
            ah[mi].h[0] = *(const v8us*)(rh);
            ah[mi].h[1] = *(const v8us*)(rh + 16);
            al[mi].h[0] = *(const v8us*)(rl);
            al[mi].h[1] = *(const v8us*)(rl + 16);
        }
#pragma unroll
        for (int ni = 0; ni < 2; ++ni) {
            const unsigned short* rh = &sBhi[(wn + ni * 16 + mr) * LSTR + bk];
            const unsigned short* rl = &sBlo[(wn + ni * 16 + mr) * LSTR + bk];
            bh[ni].h[0] = *(const v8us*)(rh);
            bh[ni].h[1] = *(const v8us*)(rh + 16);
            bl[ni].h[0] = *(const v8us*)(rl);
            bl[ni].h[1] = *(const v8us*)(rl + 16);
        }
#pragma unroll
        for (int mi = 0; mi < 4; ++mi) {
#pragma unroll
            for (int ni = 0; ni < 2; ++ni) {
                acc[mi][ni] = __builtin_amdgcn_wmma_f32_16x16x32_bf16(
                    false, ah[mi].v, false, bh[ni].v, (short)0, acc[mi][ni], false, false);
                acc[mi][ni] = __builtin_amdgcn_wmma_f32_16x16x32_bf16(
                    false, ah[mi].v, false, bl[ni].v, (short)0, acc[mi][ni], false, false);
                acc[mi][ni] = __builtin_amdgcn_wmma_f32_16x16x32_bf16(
                    false, al[mi].v, false, bh[ni].v, (short)0, acc[mi][ni], false, false);
            }
        }
    }

    // ---------------- epilogue: C/D layout -> global, add bias ----------------
#pragma unroll
    for (int ni = 0; ni < 2; ++ni) {
        const int  col = n0 + wn + ni * 16 + mr;
        const float bv = bias[col];
#pragma unroll
        for (int mi = 0; mi < 4; ++mi) {
#pragma unroll
            for (int r = 0; r < 8; ++r) {
                const int row = m0 + wm + mi * 16 + ks * 8 + r;
                out[(long long)row * NDIM + col] = acc[mi][ni][r] + bv;
            }
        }
    }
}

extern "C" void kernel_launch(void* const* d_in, const int* in_sizes, int n_in,
                              void* d_out, int out_size, void* d_ws, size_t ws_size,
                              hipStream_t stream) {
    (void)in_sizes; (void)n_in; (void)out_size; (void)d_ws; (void)ws_size;
    const float*     x    = (const float*)d_in[0];
    const float*     hw   = (const float*)d_in[1];
    const float*     bias = (const float*)d_in[2];
    const long long* rnd  = (const long long*)d_in[3];
    float*           out  = (float*)d_out;

    dim3 grid(NDIM / TILE_N, M_TOK / TILE_M);   // (64, 4)
    dim3 block(256);                            // 8 wave32s
    rz_linear_wmma_bf16x3<<<grid, block, 0, stream>>>(x, hw, bias, rnd, out);
}
